// MultiheadAttention_19499151524555
// MI455X (gfx1250) — matrix operand order repaired and checked by reading, not compiled, not run on hardware
//
#include <hip/hip_runtime.h>
#include <hip/hip_bf16.h>

typedef __bf16 bf16_t;
typedef __attribute__((ext_vector_type(16))) __bf16 v16bf;
typedef __attribute__((ext_vector_type(8)))  float  v8f;
typedef __attribute__((ext_vector_type(4)))  unsigned int v4u;
typedef __attribute__((ext_vector_type(8)))  int v8i;
typedef __attribute__((ext_vector_type(4)))  int v4i;

union FragB16 { v16bf v; uint4 u[2]; };

__device__ __forceinline__ v8f wmma_bf16(const FragB16& a, const FragB16& b, v8f c) {
    // (neg_a, A, neg_b, B, c_mod, C, reuse_a, reuse_b)
    return __builtin_amdgcn_wmma_f32_16x16x32_bf16(false, a.v, false, b.v, (short)0, c, false, false);
}

__device__ __forceinline__ v8f v8f_zero() {
    v8f z = {0.f, 0.f, 0.f, 0.f, 0.f, 0.f, 0.f, 0.f};
    return z;
}

__device__ __forceinline__ void wait_async0() {
#if __has_builtin(__builtin_amdgcn_s_wait_asynccnt)
    __builtin_amdgcn_s_wait_asynccnt(0);
#else
    asm volatile("s_wait_asynccnt 0x0" ::: "memory");
#endif
}

__device__ __forceinline__ void wait_tensor0() {
#if __has_builtin(__builtin_amdgcn_s_wait_tensorcnt)
    __builtin_amdgcn_s_wait_tensorcnt(0);
#else
    asm volatile("s_wait_tensorcnt 0x0" ::: "memory");
#endif
}

// async copy of 64 contiguous bytes: global -> LDS (ASYNCcnt path)
__device__ __forceinline__ void async_copy64(unsigned lds_addr, const void* gsrc) {
    asm volatile(
        "global_load_async_to_lds_b128 %0, %1, off\n\t"
        "global_load_async_to_lds_b128 %0, %1, off offset:16\n\t"
        "global_load_async_to_lds_b128 %0, %1, off offset:32\n\t"
        "global_load_async_to_lds_b128 %0, %1, off offset:48"
        :: "v"(lds_addr), "v"(gsrc) : "memory");
}

constexpr int   WIDTH  = 1024;
constexpr int   TCTX   = 2048;
constexpr int   HEADS  = 16;
constexpr int   CH     = 64;
constexpr float QK_SCALE = 0.35355339059327378f;  // 1 / 64^(1/4), applied to both Q and K

#if __has_builtin(__builtin_amdgcn_tensor_load_to_lds)
#define HAVE_TDM 1
// TDM load: 2D tile 64 rows x 64 bf16 (128B/row), row-major contiguous in memory,
// into LDS with 144B row stride (pad 4 DWORDs after every 32 DWORDs).
// 6-arg builtin (clang-23 / therock-10.0): (v4u g0, v8i g1, v4i g2, v4i g3, v8i, i32 cpol)
__device__ __forceinline__ void tdm_load_k_chunk(unsigned lds_addr, const void* gptr) {
    const unsigned long long ga = (unsigned long long)(size_t)gptr;
    v4u g0;
    g0.x = 1u;                                       // count=1, user descriptor
    g0.y = lds_addr;                                 // lds_addr (bytes)
    g0.z = (unsigned)(ga & 0xFFFFFFFFu);             // global_addr[31:0]
    g0.w = (unsigned)((ga >> 32) & 0x1FFFFFFu)       // global_addr[56:32]
         | (2u << 30);                               // type=2 ("image")
    v8i g1;
    g1[0] = (int)((3u << 25)                         // pad_amount   = 4 DWORDs (code 3)
                | (4u << 22)                         // pad_interval = 32 DWORDs (code 4)
                | (1u << 20)                         // pad_enable
                | (1u << 16));                       // data_size = 2 bytes
    g1[1] = (int)(64u << 16);                        // tensor_dim0 = 64 (low16 @ bits[63:48])
    g1[2] = (int)(64u << 16);                        // tensor_dim1 = 64 (low16 @ bits[95:80])
    g1[3] = (int)(64u << 16);                        // tile_dim0 = 64 (bits[127:112])
    g1[4] = 64;                                      // tile_dim1 = 64 (bits[143:128])
    g1[5] = 64;                                      // tensor_dim0_stride = 64 elems
    g1[6] = 0;
    g1[7] = 0;
    const v4i z4 = {0, 0, 0, 0};
    const v8i z8 = {0, 0, 0, 0, 0, 0, 0, 0};
    __builtin_amdgcn_tensor_load_to_lds(g0, g1, z4, z4, z8, 0);
}
#else
#define HAVE_TDM 0
#endif

// ---------------------------------------------------------------------------
// Kernel 1: QKV projection.  C[m,n] = sum_k x[m,k] * w_qkv[n,k] + b_qkv[n]
// M = 4096 (= b*2048 + t), N = 3072, K = 1024.
// Block: 256 thr (8 waves), tile 128(M) x 64(N), Kc = 64, double-buffered LDS:
// iteration i stages chunk i+1 while WMMAs consume chunk i (1 barrier/iter).
// Epilogue scatters bf16 to Q/K/V [B,H,T,C], scale folded into Q and K.
// ---------------------------------------------------------------------------
__global__ __launch_bounds__(256)
void qkv_gemm_kernel(const float* __restrict__ x, const float* __restrict__ wqkv,
                     const float* __restrict__ bqkv,
                     bf16_t* __restrict__ Qo, bf16_t* __restrict__ Ko, bf16_t* __restrict__ Vo)
{
    __shared__ bf16_t As[2][128 * 72];   // 128 rows x 64 cols + pad (144B stride)
    __shared__ bf16_t Bs[2][64 * 72];

    const int tid  = threadIdx.x;
    const int lane = tid & 31;
    const int wv   = tid >> 5;
    const int n0   = blockIdx.x * 64;
    const int m0   = blockIdx.y * 128;
    const int wm   = (wv & 3) * 32;
    const int wn   = (wv >> 2) * 32;
    const int half = lane >> 4;
    const int ln   = lane & 15;

    v8f acc[2][2];
    acc[0][0] = v8f_zero(); acc[0][1] = v8f_zero();
    acc[1][0] = v8f_zero(); acc[1][1] = v8f_zero();

    const int rb  = tid >> 4;          // 0..15
    const int col = (tid & 15) * 4;    // 0..60

    auto stage = [&](int buf, int k0) {
        #pragma unroll
        for (int i = 0; i < 8; ++i) {              // A: 128 x 64 fp32 -> bf16
            const int r = rb + i * 16;
            const float4 f = *(const float4*)(x + (size_t)(m0 + r) * WIDTH + k0 + col);
            union { uint2 u; bf16_t h[4]; } pk;
            pk.h[0] = (bf16_t)f.x; pk.h[1] = (bf16_t)f.y;
            pk.h[2] = (bf16_t)f.z; pk.h[3] = (bf16_t)f.w;
            *(uint2*)&As[buf][r * 72 + col] = pk.u;
        }
        #pragma unroll
        for (int i = 0; i < 4; ++i) {              // B: 64 x 64 fp32 -> bf16
            const int r = rb + i * 16;
            const float4 f = *(const float4*)(wqkv + (size_t)(n0 + r) * WIDTH + k0 + col);
            union { uint2 u; bf16_t h[4]; } pk;
            pk.h[0] = (bf16_t)f.x; pk.h[1] = (bf16_t)f.y;
            pk.h[2] = (bf16_t)f.z; pk.h[3] = (bf16_t)f.w;
            *(uint2*)&Bs[buf][r * 72 + col] = pk.u;
        }
    };

    stage(0, 0);
    __syncthreads();

    #pragma unroll 2
    for (int it = 0; it < WIDTH / 64; ++it) {
        const int cur = it & 1;
        if (it + 1 < WIDTH / 64) {
            stage(cur ^ 1, (it + 1) * 64);                          // in flight under WMMAs
            __builtin_prefetch(x    + (size_t)(m0 + rb) * WIDTH + (it + 2) * 64, 0, 1);
            __builtin_prefetch(wqkv + (size_t)(n0 + rb) * WIDTH + (it + 2) * 64, 0, 1);
        }
        #pragma unroll
        for (int kt = 0; kt < 2; ++kt) {
            FragB16 af[2], bfr[2];
            #pragma unroll
            for (int mt = 0; mt < 2; ++mt) {
                const int r = wm + mt * 16 + ln;                    // A: M = lane%16
                af[mt].u[0] = *(const uint4*)&As[cur][r * 72 + kt * 32 + half * 16];
                af[mt].u[1] = *(const uint4*)&As[cur][r * 72 + kt * 32 + half * 16 + 8];
            }
            #pragma unroll
            for (int nt = 0; nt < 2; ++nt) {
                const int r = wn + nt * 16 + ln;                    // B: N = lane%16
                bfr[nt].u[0] = *(const uint4*)&Bs[cur][r * 72 + kt * 32 + half * 16];
                bfr[nt].u[1] = *(const uint4*)&Bs[cur][r * 72 + kt * 32 + half * 16 + 8];
            }
            #pragma unroll
            for (int mt = 0; mt < 2; ++mt)
                #pragma unroll
                for (int nt = 0; nt < 2; ++nt)
                    acc[mt][nt] = wmma_bf16(af[mt], bfr[nt], acc[mt][nt]);
        }
        __syncthreads();
    }

    // Epilogue: each 64-wide N tile is exactly one of q/k/v for one head.
    const int which = (n0 >> 6) % 3;
    const int h     = n0 / 192;
    bf16_t* dst = (which == 0) ? Qo : (which == 1) ? Ko : Vo;
    const float sc = (which == 2) ? 1.0f : QK_SCALE;

    #pragma unroll
    for (int mt = 0; mt < 2; ++mt)
        #pragma unroll
        for (int nt = 0; nt < 2; ++nt) {
            const int c = wn + nt * 16 + ln;          // channel within head (tile width==64)
            const float bias = bqkv[n0 + c];
            #pragma unroll
            for (int v = 0; v < 8; ++v) {
                const int m = m0 + wm + mt * 16 + half * 8 + v;   // D: M = v + half*8
                const int b = m >> 11;
                const int t = m & (TCTX - 1);
                const float val = (acc[mt][nt][v] + bias) * sc;
                dst[((size_t)((b * HEADS + h) * TCTX + t)) * CH + c] = (bf16_t)val;
            }
        }
}

// ---------------------------------------------------------------------------
// Kernel 2: flash attention per (b,h).  Block = 128 thr (4 waves),
// each wave owns 16 Q rows; stream K/V in 64-row chunks through LDS.
// K chunk staged by the Tensor Data Mover (TENSORcnt) when available.
// ---------------------------------------------------------------------------
__global__ __launch_bounds__(128)
void attention_kernel(const bf16_t* __restrict__ Q, const bf16_t* __restrict__ K,
                      const bf16_t* __restrict__ V, bf16_t* __restrict__ attn)
{
    __shared__ bf16_t Ks[64 * 72];        // K rows (s) x channels, stride 72 (144B)
    __shared__ bf16_t Vt[64 * 72];        // V transposed: [c][s]
    __shared__ bf16_t Ps[4 * 16 * 72];    // per-wave P scratch (16 x 64)

    const int tid  = threadIdx.x;
    const int lane = tid & 31;
    const int wv   = tid >> 5;
    const int half = lane >> 4;
    const int ln   = lane & 15;
    const int bh   = blockIdx.y;                       // b*16 + h
    const int q0   = blockIdx.x * 64 + wv * 16;        // this wave's Q row base

    const bf16_t* Qp = Q + (size_t)bh * TCTX * CH;
    const bf16_t* Kp = K + (size_t)bh * TCTX * CH;
    const bf16_t* Vp = V + (size_t)bh * TCTX * CH;

    // Q fragments (A-matrix, M=16 rows, K=64 channels -> 2 k-steps)
    FragB16 qa[2];
    #pragma unroll
    for (int kt = 0; kt < 2; ++kt) {
        const bf16_t* p = Qp + (size_t)(q0 + ln) * CH + kt * 32 + half * 16;
        qa[kt].u[0] = *(const uint4*)p;
        qa[kt].u[1] = *(const uint4*)(p + 8);
    }

    v8f o[4];
    #pragma unroll
    for (int nt = 0; nt < 4; ++nt) o[nt] = v8f_zero();
    float m_i[8], l_i[8];
    #pragma unroll
    for (int v = 0; v < 8; ++v) { m_i[v] = -1e30f; l_i[v] = 0.f; }

    bf16_t* Pw = Ps + wv * 16 * 72;
#if HAVE_TDM
    const unsigned ldsK = (unsigned)(unsigned long long)(size_t)&Ks[0];
#endif

    for (int s0 = 0; s0 < TCTX; s0 += 64) {
        // stage K chunk
#if HAVE_TDM
        if (wv == 0) {
            tdm_load_k_chunk(ldsK, (const void*)(Kp + (size_t)s0 * CH));
            wait_tensor0();
        }
#else
        {
            const int r  = tid >> 1;
            const int hh = tid & 1;
            const uint4* src = (const uint4*)(Kp + (size_t)(s0 + r) * CH + hh * 32);
            uint4* d = (uint4*)&Ks[r * 72 + hh * 32];
            d[0] = src[0]; d[1] = src[1]; d[2] = src[2]; d[3] = src[3];
        }
#endif
        // stage V transposed
        #pragma unroll
        for (int i = 0; i < 4; ++i) {
            const int idx = tid + i * 128;
            const int s   = idx >> 3;
            const int cc  = (idx & 7) * 8;
            union { uint4 u; bf16_t h[8]; } tmp;
            tmp.u = *(const uint4*)(Vp + (size_t)(s0 + s) * CH + cc);
            #pragma unroll
            for (int j = 0; j < 8; ++j) Vt[(cc + j) * 72 + s] = tmp.h[j];
        }
        __syncthreads();

        // S tile: 16(t) x 64(s)
        v8f sacc[4];
        #pragma unroll
        for (int nt = 0; nt < 4; ++nt) sacc[nt] = v8f_zero();
        #pragma unroll
        for (int nt = 0; nt < 4; ++nt)
            #pragma unroll
            for (int kt = 0; kt < 2; ++kt) {
                FragB16 kb;   // B: N = s (lane%16), K = channel
                const bf16_t* p = &Ks[(nt * 16 + ln) * 72 + kt * 32 + half * 16];
                kb.u[0] = *(const uint4*)p;
                kb.u[1] = *(const uint4*)(p + 8);
                sacc[nt] = wmma_bf16(qa[kt], kb, sacc[nt]);
            }

        // online softmax (rows live in 16-lane halves; D layout M = v + half*8)
        float mnew[8], corr[8];
        #pragma unroll
        for (int v = 0; v < 8; ++v) {
            float mc = fmaxf(fmaxf(sacc[0][v], sacc[1][v]), fmaxf(sacc[2][v], sacc[3][v]));
            mc = fmaxf(mc, __shfl_xor(mc, 1, 32));
            mc = fmaxf(mc, __shfl_xor(mc, 2, 32));
            mc = fmaxf(mc, __shfl_xor(mc, 4, 32));
            mc = fmaxf(mc, __shfl_xor(mc, 8, 32));
            mnew[v] = fmaxf(m_i[v], mc);
        }
        #pragma unroll
        for (int nt = 0; nt < 4; ++nt)
            #pragma unroll
            for (int v = 0; v < 8; ++v)
                sacc[nt][v] = __expf(sacc[nt][v] - mnew[v]);
        #pragma unroll
        for (int v = 0; v < 8; ++v) {
            float r = sacc[0][v] + sacc[1][v] + sacc[2][v] + sacc[3][v];
            r += __shfl_xor(r, 1, 32);
            r += __shfl_xor(r, 2, 32);
            r += __shfl_xor(r, 4, 32);
            r += __shfl_xor(r, 8, 32);
            corr[v] = __expf(m_i[v] - mnew[v]);
            l_i[v]  = l_i[v] * corr[v] + r;
            m_i[v]  = mnew[v];
        }
        #pragma unroll
        for (int nt = 0; nt < 4; ++nt)
            #pragma unroll
            for (int v = 0; v < 8; ++v)
                o[nt][v] *= corr[v];

        // P (D layout) -> LDS -> A-fragment layout
        #pragma unroll
        for (int nt = 0; nt < 4; ++nt)
            #pragma unroll
            for (int v = 0; v < 8; ++v)
                Pw[(half * 8 + v) * 72 + nt * 16 + ln] = (bf16_t)sacc[nt][v];

        // O += P @ V
        #pragma unroll
        for (int kt = 0; kt < 2; ++kt) {
            FragB16 pa;
            const bf16_t* p = &Pw[ln * 72 + kt * 32 + half * 16];
            pa.u[0] = *(const uint4*)p;
            pa.u[1] = *(const uint4*)(p + 8);
            #pragma unroll
            for (int nt = 0; nt < 4; ++nt) {
                FragB16 vb;   // B: N = c, K = s (contiguous thanks to transpose)
                const bf16_t* q = &Vt[(nt * 16 + ln) * 72 + kt * 32 + half * 16];
                vb.u[0] = *(const uint4*)q;
                vb.u[1] = *(const uint4*)(q + 8);
                o[nt] = wmma_bf16(pa, vb, o[nt]);
            }
        }
        __syncthreads();
    }

    // write attn [b, t, h*64 + c] as bf16
    const int b = bh >> 4;
    const int h = bh & 15;
    #pragma unroll
    for (int nt = 0; nt < 4; ++nt)
        #pragma unroll
        for (int v = 0; v < 8; ++v) {
            const int t = q0 + half * 8 + v;
            const int c = nt * 16 + ln;
            attn[((size_t)(b * TCTX + t)) * WIDTH + h * CH + c] = (bf16_t)(o[nt][v] / l_i[v]);
        }
}

// ---------------------------------------------------------------------------
// Kernel 3: output projection. out[m,n] = sum_k attn[m,k]*w_proj[n,k] + b_proj[n]
// attn source is already bf16 -> A staged via async global->LDS (ASYNCcnt),
// double-buffered so the async DMA overlaps the WMMA stream.
// ---------------------------------------------------------------------------
__global__ __launch_bounds__(256)
void proj_gemm_kernel(const bf16_t* __restrict__ a, const float* __restrict__ wproj,
                      const float* __restrict__ bproj, float* __restrict__ out)
{
    __shared__ bf16_t As[2][128 * 72];
    __shared__ bf16_t Bs[2][64 * 72];

    const int tid  = threadIdx.x;
    const int lane = tid & 31;
    const int wv   = tid >> 5;
    const int n0   = blockIdx.x * 64;
    const int m0   = blockIdx.y * 128;
    const int wm   = (wv & 3) * 32;
    const int wn   = (wv >> 2) * 32;
    const int half = lane >> 4;
    const int ln   = lane & 15;

    v8f acc[2][2];
    acc[0][0] = v8f_zero(); acc[0][1] = v8f_zero();
    acc[1][0] = v8f_zero(); acc[1][1] = v8f_zero();

    const int rb  = tid >> 4;
    const int col = (tid & 15) * 4;

    auto stage = [&](int buf, int k0) {
        // A tile: 128 rows x 64 bf16, pure copy -> async-to-LDS
        const int r  = tid >> 1;          // 0..127
        const int hh = tid & 1;           // half-row of 32 elems (64B)
        const unsigned lbase = (unsigned)(unsigned long long)(size_t)&As[buf][r * 72 + hh * 32];
        async_copy64(lbase, (const void*)(a + (size_t)(m0 + r) * WIDTH + k0 + hh * 32));
        // B tile: w_proj fp32 -> bf16
        #pragma unroll
        for (int i = 0; i < 4; ++i) {
            const int rr = rb + i * 16;
            const float4 f = *(const float4*)(wproj + (size_t)(n0 + rr) * WIDTH + k0 + col);
            union { uint2 u; bf16_t h[4]; } pk;
            pk.h[0] = (bf16_t)f.x; pk.h[1] = (bf16_t)f.y;
            pk.h[2] = (bf16_t)f.z; pk.h[3] = (bf16_t)f.w;
            *(uint2*)&Bs[buf][rr * 72 + col] = pk.u;
        }
    };

    stage(0, 0);
    wait_async0();
    __syncthreads();

    #pragma unroll 2
    for (int it = 0; it < WIDTH / 64; ++it) {
        const int cur = it & 1;
        if (it + 1 < WIDTH / 64) {
            stage(cur ^ 1, (it + 1) * 64);                          // async DMA under WMMAs
            __builtin_prefetch(wproj + (size_t)(n0 + rb) * WIDTH + (it + 2) * 64, 0, 1);
        }
        #pragma unroll
        for (int kt = 0; kt < 2; ++kt) {
            FragB16 af[2], bfr[2];
            #pragma unroll
            for (int mt = 0; mt < 2; ++mt) {
                const int r = wm + mt * 16 + ln;
                af[mt].u[0] = *(const uint4*)&As[cur][r * 72 + kt * 32 + half * 16];
                af[mt].u[1] = *(const uint4*)&As[cur][r * 72 + kt * 32 + half * 16 + 8];
            }
            #pragma unroll
            for (int nt = 0; nt < 2; ++nt) {
                const int r = wn + nt * 16 + ln;
                bfr[nt].u[0] = *(const uint4*)&Bs[cur][r * 72 + kt * 32 + half * 16];
                bfr[nt].u[1] = *(const uint4*)&Bs[cur][r * 72 + kt * 32 + half * 16 + 8];
            }
            #pragma unroll
            for (int mt = 0; mt < 2; ++mt)
                #pragma unroll
                for (int nt = 0; nt < 2; ++nt)
                    acc[mt][nt] = wmma_bf16(af[mt], bfr[nt], acc[mt][nt]);
        }
        wait_async0();
        __syncthreads();
    }

    #pragma unroll
    for (int mt = 0; mt < 2; ++mt)
        #pragma unroll
        for (int nt = 0; nt < 2; ++nt) {
            const int c = wn + nt * 16 + ln;
            const float bias = bproj[n0 + c];
            #pragma unroll
            for (int v = 0; v < 8; ++v) {
                const int m = m0 + wm + mt * 16 + half * 8 + v;
                out[(size_t)m * WIDTH + n0 + c] = acc[mt][nt][v] + bias;
            }
        }
}

// ---------------------------------------------------------------------------
extern "C" void kernel_launch(void* const* d_in, const int* in_sizes, int n_in,
                              void* d_out, int out_size, void* d_ws, size_t ws_size,
                              hipStream_t stream) {
    (void)in_sizes; (void)n_in; (void)out_size; (void)ws_size;
    const float* x     = (const float*)d_in[0];
    const float* wqkv  = (const float*)d_in[1];
    const float* bqkv  = (const float*)d_in[2];
    const float* wproj = (const float*)d_in[3];
    const float* bproj = (const float*)d_in[4];
    float* out = (float*)d_out;

    // workspace: Q,K,V bf16 [B,H,T,C] + attn bf16 [B,T,W]  (3*8MB + 8MB = 32MB)
    const size_t qkv_elems = (size_t)2 * HEADS * TCTX * CH;   // 4,194,304
    bf16_t* Qb   = (bf16_t*)d_ws;
    bf16_t* Kb   = Qb + qkv_elems;
    bf16_t* Vb   = Kb + qkv_elems;
    bf16_t* attn = Vb + qkv_elems;

    qkv_gemm_kernel<<<dim3(3 * WIDTH / 64, 4096 / 128), 256, 0, stream>>>(
        x, wqkv, bqkv, Qb, Kb, Vb);
    attention_kernel<<<dim3(TCTX / 64, 2 * HEADS), 128, 0, stream>>>(
        Qb, Kb, Vb, attn);
    proj_gemm_kernel<<<dim3(WIDTH / 64, 4096 / 128), 256, 0, stream>>>(
        attn, wproj, bproj, out);
}
